// EntitiesIndexingHeadRuleBased_31181462569594
// MI455X (gfx1250) — matrix operand order repaired and checked by reading, not compiled, not run on hardware
//
#include <hip/hip_runtime.h>
#include <hip/hip_bf16.h>
#include <math.h>

// ---------------------------------------------------------------------------
// Problem constants (from the reference)
// ---------------------------------------------------------------------------
#define BATCH 64
#define NE    500          // entities per image
#define NR    500          // relations per image
#define C1    151          // classes + 1 (softmax width)
#define CQ    150          // classes kept after dropping last
#define KP    160          // K padded to 5 * 32 for v_wmma_f32_16x16x32_f16

typedef _Float16 v16h __attribute__((ext_vector_type(16)));
typedef _Float16 v8h  __attribute__((ext_vector_type(8)));
typedef float    v8f  __attribute__((ext_vector_type(8)));

// ---------------------------------------------------------------------------
// wave32 reductions
// ---------------------------------------------------------------------------
__device__ __forceinline__ float wave_max(float v) {
#pragma unroll
    for (int m = 16; m >= 1; m >>= 1) v = fmaxf(v, __shfl_xor(v, m, 32));
    return v;
}
__device__ __forceinline__ float wave_sum(float v) {
#pragma unroll
    for (int m = 16; m >= 1; m >>= 1) v += __shfl_xor(v, m, 32);
    return v;
}

// ---------------------------------------------------------------------------
// Softmax over 151 logits by one wave; drop last class, store 150 f16 probs
// (zero-padded to KP=160) and return sum of squared probs (broadcast to wave).
// ---------------------------------------------------------------------------
__device__ __forceinline__ float softmax_row_store(const float* __restrict__ lg,
                                                   _Float16* __restrict__ row,
                                                   int lane) {
    float v[5];
    float mx = -3.4e38f;
#pragma unroll
    for (int j = 0; j < 5; ++j) {
        int i = lane + 32 * j;
        v[j] = (i < C1) ? lg[i] : -3.4e38f;
        mx = fmaxf(mx, v[j]);
    }
    mx = wave_max(mx);
    float s = 0.f;
#pragma unroll
    for (int j = 0; j < 5; ++j) {
        int i = lane + 32 * j;
        v[j] = (i < C1) ? expf(v[j] - mx) : 0.f;
        s += v[j];
    }
    s = wave_sum(s);
    const float inv = 1.f / s;
    float ssq = 0.f;
#pragma unroll
    for (int j = 0; j < 5; ++j) {
        int i = lane + 32 * j;                // covers 0..159 exactly
        if (i < CQ) {
            float p = v[j] * inv;
            row[i] = (_Float16)p;
            ssq += p * p;
        } else {
            row[i] = (_Float16)0.f;           // K padding -> contributes 0 to dot
        }
    }
    return wave_sum(ssq);
}

// ---------------------------------------------------------------------------
// Stage 1a: entity preprocessing. One wave per (image, entity).
// Writes: f16 prob row [KP], params row [x0 y0 x1 y1 cx*W cy*H score ssq].
// ---------------------------------------------------------------------------
__global__ void ent_prep_kernel(const float* __restrict__ boxes,
                                const float* __restrict__ logits,
                                const float* __restrict__ tsz,
                                _Float16* __restrict__ bent,
                                float* __restrict__ epar) {
    const int lane = threadIdx.x & 31;
    const int wid  = blockIdx.x * (blockDim.x >> 5) + (threadIdx.x >> 5);
    if (wid >= BATCH * NE) return;
    const int b = wid / NE;

    // softmax + f16 store + sum(p^2)
    const float ssq = softmax_row_store(logits + (size_t)wid * C1,
                                        bent + (size_t)wid * KP, lane);
    // score = max prob over kept classes
    float sc = 0.f;
    {
        const _Float16* row = bent + (size_t)wid * KP;
#pragma unroll
        for (int j = 0; j < 5; ++j) {
            int i = lane + 32 * j;
            if (i < CQ) sc = fmaxf(sc, (float)row[i]);
        }
        sc = wave_max(sc);
    }
    if (lane == 0) {
        const float H = tsz[b * 2 + 0], W = tsz[b * 2 + 1];
        const float* bx = boxes + (size_t)wid * 4;
        const float cx = bx[0], cy = bx[1], w = bx[2], h = bx[3];
        float* p = epar + (size_t)wid * 8;
        p[0] = (cx - 0.5f * w) * W;
        p[1] = (cy - 0.5f * h) * H;
        p[2] = (cx + 0.5f * w) * W;
        p[3] = (cy + 0.5f * h) * H;
        p[4] = cx * W;                        // scaled box center == (cx*W, cy*H)
        p[5] = cy * H;
        p[6] = sc;
        p[7] = ssq;
    }
}

// ---------------------------------------------------------------------------
// Stage 1b: relation preprocessing. One wave per (image, relation).
// Writes f16 rows for rs/ro softmaxes and a 16-float param row:
// [rs_box xyxy, ro_box xyxy, vec xyzw, ssq_rs, ssq_ro, 0, 0] (all scaled).
// ---------------------------------------------------------------------------
__global__ void rel_prep_kernel(const float* __restrict__ rs_lg,
                                const float* __restrict__ ro_lg,
                                const float* __restrict__ rs_bx,
                                const float* __restrict__ ro_bx,
                                const float* __restrict__ vec,
                                const float* __restrict__ tsz,
                                _Float16* __restrict__ asub,
                                _Float16* __restrict__ aobj,
                                float* __restrict__ rpar) {
    const int lane = threadIdx.x & 31;
    const int wid  = blockIdx.x * (blockDim.x >> 5) + (threadIdx.x >> 5);
    if (wid >= BATCH * NR) return;
    const int b = wid / NR;

    const float ssq_rs = softmax_row_store(rs_lg + (size_t)wid * C1,
                                           asub + (size_t)wid * KP, lane);
    const float ssq_ro = softmax_row_store(ro_lg + (size_t)wid * C1,
                                           aobj + (size_t)wid * KP, lane);
    if (lane == 0) {
        const float H = tsz[b * 2 + 0], W = tsz[b * 2 + 1];
        float* p = rpar + (size_t)wid * 16;
        const float* s = rs_bx + (size_t)wid * 4;
        const float* o = ro_bx + (size_t)wid * 4;
        const float* v = vec   + (size_t)wid * 4;
        p[0]  = (s[0] - 0.5f * s[2]) * W;
        p[1]  = (s[1] - 0.5f * s[3]) * H;
        p[2]  = (s[0] + 0.5f * s[2]) * W;
        p[3]  = (s[1] + 0.5f * s[3]) * H;
        p[4]  = (o[0] - 0.5f * o[2]) * W;
        p[5]  = (o[1] - 0.5f * o[3]) * H;
        p[6]  = (o[0] + 0.5f * o[2]) * W;
        p[7]  = (o[1] + 0.5f * o[3]) * H;
        p[8]  = v[0] * W;
        p[9]  = v[1] * H;
        p[10] = v[2] * W;
        p[11] = v[3] * H;
        p[12] = ssq_rs;
        p[13] = ssq_ro;
        p[14] = 0.f;
        p[15] = 0.f;
    }
}

// ---------------------------------------------------------------------------
// WMMA fragment loaders (CDNA5 wave32 layouts, CDNA5 ISA §7.12.2).
// A 16x32 f16: lanes 0-15 = rows M0..15 holding K 0..7 (v0..3) + 16..23 (v4..7);
//              lanes 16-31 = same rows holding K 8..15 + 24..31.
// B 32x16 f16: lane = column; same K split as A (K 0..7 + 16..23 / 8..15 + 24..31).
// Both operands here are rows of a row-major [rows x KP] f16 matrix.
// ---------------------------------------------------------------------------
__device__ __forceinline__ v16h load_frag_a(const _Float16* __restrict__ row,
                                            int k0, int h) {
    union { v16h v; v8h p[2]; } u;
    u.p[0] = *(const v8h*)(row + k0 + h * 8);
    u.p[1] = *(const v8h*)(row + k0 + 16 + h * 8);
    return u.v;
}
__device__ __forceinline__ v16h load_frag_b(const _Float16* __restrict__ row,
                                            int k0, int h) {
    union { v16h v; v8h p[2]; } u;
    const _Float16* p = row + k0 + h * 8;
    u.p[0] = *(const v8h*)(p);
    u.p[1] = *(const v8h*)(p + 16);
    return u.v;
}

// GIoU(entity box a, relation box b), clipped at 0, matching the reference.
__device__ __forceinline__ float giou_clip(float ax0, float ay0, float ax1, float ay1,
                                           float bx0, float by0, float bx1, float by1) {
    const float areaA = (ax1 - ax0) * (ay1 - ay0);
    const float areaB = (bx1 - bx0) * (by1 - by0);
    const float iw = fminf(ax1, bx1) - fmaxf(ax0, bx0);
    const float ih = fminf(ay1, by1) - fmaxf(ay0, by0);
    const float inter = fmaxf(iw, 0.f) * fmaxf(ih, 0.f);
    const float uni = areaA + areaB - inter;
    const float cw = fmaxf(ax1, bx1) - fminf(ax0, bx0);
    const float ch = fmaxf(ay1, by1) - fminf(ay0, by0);
    const float ac = fmaxf(cw, 0.f) * fmaxf(ch, 0.f);
    const float g = inter / uni - (ac - uni) / ac;
    return fmaxf(g, 0.f);
}

// ---------------------------------------------------------------------------
// Stage 2: fused WMMA + epilogue. Block = 256 threads = 8 waves covering a
// 64(rel) x 32(ent) tile; each wave owns one 16x16 tile with TWO f32
// accumulators (sub & obj) sharing the B (ent_prob) fragment. K-loop: 5 x 32.
// Epilogue applies cdist (ssq_r + ssq_e - 2*dot), L1 center match, GIoU, and
// writes each output element exactly once (outputs are the bandwidth limit).
// ---------------------------------------------------------------------------
__global__ void match_kernel(const _Float16* __restrict__ asub,
                             const _Float16* __restrict__ aobj,
                             const _Float16* __restrict__ bent,
                             const float* __restrict__ epar,
                             const float* __restrict__ rpar,
                             float* __restrict__ outS,
                             float* __restrict__ outO) {
    const int lane = threadIdx.x & 31;
    const int w    = threadIdx.x >> 5;      // wave 0..7
    const int wm   = w >> 1;                // 0..3 : 16-row slab within tile
    const int wn   = w & 1;                 // 0..1 : 16-col slab within tile
    const int b    = blockIdx.z;
    const int RM   = blockIdx.y * 64 + wm * 16;   // relation row base
    const int EN   = blockIdx.x * 32 + wn * 16;   // entity col base
    const int h    = lane >> 4;

    int ar = RM + (lane & 15); if (ar > NR - 1) ar = NR - 1;   // clamp: padded
    int br = EN + (lane & 15); if (br > NE - 1) br = NE - 1;   // rows are valid

    const _Float16* aS = asub + ((size_t)b * NR + ar) * KP;
    const _Float16* aO = aobj + ((size_t)b * NR + ar) * KP;
    const _Float16* bE = bent + ((size_t)b * NE + br) * KP;

    v8f accS = {};
    v8f accO = {};
#pragma unroll
    for (int k0 = 0; k0 < KP; k0 += 32) {
        if (k0 + 32 < KP) {                 // gfx1250 global_prefetch_b8 path
            __builtin_prefetch(bE + k0 + 32, 0, 3);
            __builtin_prefetch(aS + k0 + 32, 0, 3);
        }
        const v16h fa_s = load_frag_a(aS, k0, h);
        const v16h fa_o = load_frag_a(aO, k0, h);
        const v16h fb   = load_frag_b(bE, k0, h);
        accS = __builtin_amdgcn_wmma_f32_16x16x32_f16(false, fa_s, false, fb,
                                                      (short)0, accS, false, false);
        accO = __builtin_amdgcn_wmma_f32_16x16x32_f16(false, fa_o, false, fb,
                                                      (short)0, accO, false, false);
    }

    // Epilogue. C/D layout: lane = N + 16*(M>=8), VGPR r holds M = (lane>>4)*8 + r.
    const int e = EN + (lane & 15);
    if (e >= NE) return;
    const float4 e0 = *(const float4*)(epar + ((size_t)b * NE + e) * 8);     // box xyxy
    const float4 e1 = *(const float4*)(epar + ((size_t)b * NE + e) * 8 + 4); // cx cy score ssq
#pragma unroll
    for (int r = 0; r < 8; ++r) {
        const int rel = RM + h * 8 + r;
        if (rel >= NR) continue;
        const float* rp = rpar + ((size_t)b * NR + rel) * 16;
        const float4 rs = *(const float4*)(rp);
        const float4 ro = *(const float4*)(rp + 4);
        const float4 vc = *(const float4*)(rp + 8);
        const float ssq_rs = rp[12], ssq_ro = rp[13];
        const size_t oidx = ((size_t)b * NR + rel) * NE + e;

        // --- subject head ---
        {
            const float d2  = ssq_rs + e1.w - 2.f * accS[r];
            const float cls = 1.f / (sqrtf(fmaxf(d2, 1e-12f)) + 1.f);
            const float dst = fabsf(vc.x - e1.x) + fabsf(vc.y - e1.y);
            const float m   = e1.z / (dst + 1.f);
            const float g   = giou_clip(e0.x, e0.y, e0.z, e0.w,
                                        rs.x, rs.y, rs.z, rs.w);
            outS[oidx] = m * cls * g;
        }
        // --- object head ---
        {
            const float d2  = ssq_ro + e1.w - 2.f * accO[r];
            const float cls = 1.f / (sqrtf(fmaxf(d2, 1e-12f)) + 1.f);
            const float dst = fabsf(vc.z - e1.x) + fabsf(vc.w - e1.y);
            const float m   = e1.z / (dst + 1.f);
            const float g   = giou_clip(e0.x, e0.y, e0.z, e0.w,
                                        ro.x, ro.y, ro.z, ro.w);
            outO[oidx] = m * cls * g;
        }
    }
}

// ---------------------------------------------------------------------------
// Launch. Workspace layout (all chunks 256B-aligned by construction):
//   A_sub f16 [B][NR][KP]   10.24 MB
//   A_obj f16 [B][NR][KP]   10.24 MB
//   B_ent f16 [B][NE][KP]   10.24 MB
//   epar  f32 [B][NE][8]     1.02 MB
//   rpar  f32 [B][NR][16]    2.05 MB   -> total ~33.8 MB of d_ws
// ---------------------------------------------------------------------------
extern "C" void kernel_launch(void* const* d_in, const int* in_sizes, int n_in,
                              void* d_out, int out_size, void* d_ws, size_t ws_size,
                              hipStream_t stream) {
    (void)in_sizes; (void)n_in; (void)out_size; (void)ws_size;
    const float* boxes     = (const float*)d_in[0];
    const float* logits    = (const float*)d_in[1];
    const float* ro_logits = (const float*)d_in[2];
    const float* rs_logits = (const float*)d_in[3];
    const float* ro_box    = (const float*)d_in[4];
    const float* rs_box    = (const float*)d_in[5];
    const float* rel_vec   = (const float*)d_in[6];
    const float* tsz       = (const float*)d_in[7];

    char* ws = (char*)d_ws;
    const size_t szF16 = (size_t)BATCH * NR * KP * sizeof(_Float16); // 10,240,000 B
    _Float16* Asub = (_Float16*)(ws);
    _Float16* Aobj = (_Float16*)(ws + szF16);
    _Float16* Bent = (_Float16*)(ws + 2 * szF16);
    float*    epar = (float*)   (ws + 3 * szF16);
    float*    rpar = (float*)   (ws + 3 * szF16 + (size_t)BATCH * NE * 8 * sizeof(float));

    float* outS = (float*)d_out;
    float* outO = outS + (size_t)BATCH * NR * NE;

    // 8 waves per 256-thread block; B*NE = B*NR = 32000 rows -> 4000 blocks.
    ent_prep_kernel<<<dim3((BATCH * NE) / 8), 256, 0, stream>>>(
        boxes, logits, tsz, Bent, epar);
    rel_prep_kernel<<<dim3((BATCH * NR) / 8), 256, 0, stream>>>(
        rs_logits, ro_logits, rs_box, ro_box, rel_vec, tsz, Asub, Aobj, rpar);

    // Tile grid: x = ceil(NE/32)=16, y = ceil(NR/64)=8, z = B.
    match_kernel<<<dim3(16, 8, BATCH), 256, 0, stream>>>(
        Asub, Aobj, Bent, epar, rpar, outS, outO);
}
